// BMoIE_75144747811124
// MI455X (gfx1250) — matrix operand order repaired and checked by reading, not compiled, not run on hardware
//
#include <hip/hip_runtime.h>

// ---------------------------------------------------------------------------
// MoIE stack for MI455X (gfx1250, wave32, WMMA bf16 16x16x32, f32 accumulate)
// B=16384, D=H=512, K=8 experts, 4 layers (ReLU on first 3).
//
// Per layer: Out = act( sum_e alpha[:,e] * (Act @ W[e]) + sum_e alpha[:,e]*b[e] )
// A tile converted to bf16 ONCE into LDS (reused by all 8 experts);
// alpha applied in f32 on per-expert partial WMMA accumulators;
// B fragments ping-pong double-buffered (no register copies).
// ---------------------------------------------------------------------------

typedef __attribute__((ext_vector_type(16))) __bf16 v16bf;
typedef __attribute__((ext_vector_type(8)))  __bf16 v8bf;
typedef __attribute__((ext_vector_type(4)))  __bf16 v4bf;
typedef __attribute__((ext_vector_type(8)))  float  v8f;

#define B_TOT 16384
#define DDIM  512
#define HDIM  512
#define KEXP  8
#define BM    128
#define BN    64
#define LDS_STRIDE (DDIM + 8)   // +8 bf16 pad: row stride 260 dwords == 4 mod 64

// f32 -> bf16 round-to-nearest-even (pure integer ops)
__device__ __forceinline__ __bf16 f2bf(float f) {
    unsigned int u = __builtin_bit_cast(unsigned int, f);
    u = (u + 0x7FFFu + ((u >> 16) & 1u)) >> 16;
    unsigned short s = (unsigned short)u;
    return __builtin_bit_cast(__bf16, s);
}

__device__ __forceinline__ v16bf load_a_frag(const __bf16* p) {
    const v8bf lo = *(const v8bf*)(p);
    const v8bf hi = *(const v8bf*)(p + 16);
    return __builtin_shufflevector(lo, hi, 0, 1, 2, 3, 4, 5, 6, 7,
                                           8, 9, 10, 11, 12, 13, 14, 15);
}

// ---------------------------------------------------------------------------
// Gating: alpha[b,k] = softmax_k( x[b,:] @ gate_w[:,k] + gate_b[k] )
// ---------------------------------------------------------------------------
__global__ __launch_bounds__(256)
void gate_kernel(const float* __restrict__ x,
                 const float* __restrict__ gw,   // [D, K]
                 const float* __restrict__ gb,   // [K]
                 float* __restrict__ alpha)      // [B, K]
{
    const int wave = threadIdx.x >> 5;
    const int lane = threadIdx.x & 31;
    const int b    = blockIdx.x * 8 + wave;
    const float* xr = x + (size_t)b * DDIM;

    float acc[KEXP];
#pragma unroll
    for (int k = 0; k < KEXP; ++k) acc[k] = 0.f;

    for (int i = lane; i < DDIM; i += 32) {
        const float xv = xr[i];
        const float* g = gw + (size_t)i * KEXP;
#pragma unroll
        for (int k = 0; k < KEXP; ++k) acc[k] += xv * g[k];
    }
#pragma unroll
    for (int m = 16; m >= 1; m >>= 1) {
#pragma unroll
        for (int k = 0; k < KEXP; ++k)
            acc[k] += __shfl_xor(acc[k], m, 32);
    }
    float mx = -3.4e38f;
#pragma unroll
    for (int k = 0; k < KEXP; ++k) { acc[k] += gb[k]; mx = fmaxf(mx, acc[k]); }
    float s = 0.f;
#pragma unroll
    for (int k = 0; k < KEXP; ++k) { acc[k] = __expf(acc[k] - mx); s += acc[k]; }
    const float inv = 1.f / s;
    if (lane < KEXP) alpha[(size_t)b * KEXP + lane] = acc[lane] * inv;
}

// ---------------------------------------------------------------------------
// Weight conversion f32 -> bf16 (once, amortized over 4 layers)
// ---------------------------------------------------------------------------
__global__ __launch_bounds__(256)
void cvt_bf16_kernel(const float* __restrict__ in, __bf16* __restrict__ out)
{
    const size_t i = ((size_t)blockIdx.x * blockDim.x + threadIdx.x) * 4;
    const float4 v = *(const float4*)(in + i);
    v4bf o;
    o[0] = f2bf(v.x); o[1] = f2bf(v.y); o[2] = f2bf(v.z); o[3] = f2bf(v.w);
    *(v4bf*)(out + i) = o;
}

// ---------------------------------------------------------------------------
// One MoIE layer. Workgroup tile 128x64, 8 waves; each wave: 16 rows x 64 cols
// = 4 accumulators of V_WMMA_F32_16X16X32_BF16.
// A-frag (ISA 16-bit A 16x32): lanes 0-15 -> K {0..7,16..23},
//                              lanes 16-31 -> K {8..15,24..31}.
// B-frag: lane%16 = N column; K order same as A-frag (lanes 0-15 {0..7,16..23}, 16-31 {8..15,24..31}).
// ---------------------------------------------------------------------------
__global__ __launch_bounds__(256)
void moie_layer_kernel(const float*  __restrict__ Act,   // [B, D] f32
                       const __bf16* __restrict__ W,     // [K, D, H] bf16
                       const float*  __restrict__ bias,  // [K, H]
                       const float*  __restrict__ alpha, // [B, K]
                       float*        __restrict__ Out,   // [B, H]
                       int relu)
{
    __shared__ __bf16 ldsA[BM][LDS_STRIDE];   // 128 x 520 bf16 = 130 KB

    const int tid  = threadIdx.x;
    const int wave = tid >> 5;
    const int lane = tid & 31;
    const int half = lane >> 4;     // 0: lanes 0-15, 1: lanes 16-31
    const int l15  = lane & 15;
    const int n0   = blockIdx.y * BN;

    // ---- stage A tile into LDS as bf16 (once; reused by all 8 experts) ----
    {
        const int r  = tid >> 1;            // 0..127
        const int c0 = (tid & 1) * 256;     // 0 or 256
        const float* src = Act + (size_t)(blockIdx.x * BM + r) * DDIM + c0;
#pragma unroll 4
        for (int c = 0; c < 256; c += 8) {
            const float4 f0 = *(const float4*)(src + c);
            const float4 f1 = *(const float4*)(src + c + 4);
            v8bf o;
            o[0] = f2bf(f0.x); o[1] = f2bf(f0.y); o[2] = f2bf(f0.z); o[3] = f2bf(f0.w);
            o[4] = f2bf(f1.x); o[5] = f2bf(f1.y); o[6] = f2bf(f1.z); o[7] = f2bf(f1.w);
            *(v8bf*)(&ldsA[r][c0 + c]) = o;
        }
    }
    __syncthreads();

    // C layout rows for this lane: m = rowbase + r, r = 0..7
    const int rowbase = blockIdx.x * BM + wave * 16 + half * 8;

    v8f acc[4] = {v8f{}, v8f{}, v8f{}, v8f{}};
    const int arow = wave * 16 + l15;       // A-frag row within LDS tile

    for (int e = 0; e < KEXP; ++e) {
        // Barrier: keeps all 8 waves on the same expert (B tile stays hot in
        // WGP$), and fences LDS so A-frag ds_loads can't be hoisted/piled up.
        __syncthreads();

        const __bf16* wK = W + (size_t)e * DDIM * HDIM + n0 + l15;
        auto ldB = [&](const __bf16* p) {
            v16bf v;
#pragma unroll
            for (int s = 0; s < 16; ++s)
                v[s] = p[(size_t)((s < 8 ? s : s + 8) + half * 8) * HDIM];
            return v;
        };
        if (e + 1 < KEXP)
            __builtin_prefetch(wK + (size_t)DDIM * HDIM, 0, 0);

        v8f accP[4] = {v8f{}, v8f{}, v8f{}, v8f{}};

        // prime ping-pong: bufA <- B(d0=0)
        v16bf bufA[4], bufB[4];
#pragma unroll
        for (int j = 0; j < 4; ++j)
            bufA[j] = ldB(wK + j * 16);

        // two k-steps (64 d-columns) per iteration; no register copies
#pragma unroll 1
        for (int d0 = 0; d0 < DDIM; d0 += 64) {
            // issue loads for d0+32 into bufB
            const __bf16* bn1 = wK + (size_t)(d0 + 32) * HDIM;
#pragma unroll
            for (int j = 0; j < 4; ++j)
                bufB[j] = ldB(bn1 + j * 16);

            // WMMAs on bufA (loaded one buffer ago)
            const v16bf a0 = load_a_frag(&ldsA[arow][d0 + half * 8]);
#pragma unroll
            for (int j = 0; j < 4; ++j)
                accP[j] = __builtin_amdgcn_wmma_f32_16x16x32_bf16(
                    false, a0, false, bufA[j], (short)0, accP[j], false, false);

            // issue loads for d0+64 into bufA
            if (d0 + 64 < DDIM) {
                const __bf16* bn2 = wK + (size_t)(d0 + 64) * HDIM;
#pragma unroll
                for (int j = 0; j < 4; ++j)
                    bufA[j] = ldB(bn2 + j * 16);
            }

            // WMMAs on bufB
            const v16bf a1 = load_a_frag(&ldsA[arow][d0 + 32 + half * 8]);
#pragma unroll
            for (int j = 0; j < 4; ++j)
                accP[j] = __builtin_amdgcn_wmma_f32_16x16x32_bf16(
                    false, a1, false, bufB[j], (short)0, accP[j], false, false);
        }

        // fold alpha (f32) into the running accumulator; per C-layout row
        float ar[8];
#pragma unroll
        for (int r = 0; r < 8; ++r)
            ar[r] = alpha[(size_t)(rowbase + r) * KEXP + e];
#pragma unroll
        for (int j = 0; j < 4; ++j)
#pragma unroll
            for (int r = 0; r < 8; ++r)
                acc[j][r] += ar[r] * accP[j][r];
    }

    // ---- epilogue: mixed bias, optional ReLU, f32 store ----
#pragma unroll
    for (int r = 0; r < 8; ++r) {
        const int m = rowbase + r;
        const float* al = alpha + (size_t)m * KEXP;
#pragma unroll
        for (int j = 0; j < 4; ++j) {
            const int n = n0 + l15 + j * 16;
            float be = 0.f;
#pragma unroll
            for (int k = 0; k < KEXP; ++k)
                be += al[k] * bias[(size_t)k * HDIM + n];
            float v = acc[j][r] + be;
            if (relu) v = fmaxf(v, 0.f);
            Out[(size_t)m * HDIM + n] = v;
        }
    }
}

// ---------------------------------------------------------------------------
extern "C" void kernel_launch(void* const* d_in, const int* in_sizes, int n_in,
                              void* d_out, int out_size, void* d_ws, size_t ws_size,
                              hipStream_t stream) {
    const float* x      = (const float*)d_in[0];
    const float* gate_w = (const float*)d_in[1];
    const float* gate_b = (const float*)d_in[2];
    const float* w0     = (const float*)d_in[3];
    const float* b0     = (const float*)d_in[4];
    const float* w1     = (const float*)d_in[5];
    const float* b1     = (const float*)d_in[6];
    const float* w2     = (const float*)d_in[7];
    const float* b2     = (const float*)d_in[8];
    const float* wo     = (const float*)d_in[9];
    const float* bo     = (const float*)d_in[10];

    // workspace layout
    char* ws = (char*)d_ws;
    float*  alpha = (float*)ws;                                   // 512 KB
    __bf16* wbf   = (__bf16*)(ws + (512 << 10));                  // 16 MB
    float*  hA    = (float*)(ws + (512 << 10) + (16 << 20));      // 32 MB
    float*  hB    = (float*)((char*)hA + (32 << 20));             // 32 MB

    // 1) gating -> alpha [B,8]
    gate_kernel<<<B_TOT / 8, 256, 0, stream>>>(x, gate_w, gate_b, alpha);

    // 2) weights f32 -> bf16 (each [K,D,H] = 2,097,152 elems; 4 elems/thread)
    const size_t WSZ = (size_t)KEXP * DDIM * HDIM;
    const int cvt_grid = (int)(WSZ / 4 / 256);
    cvt_bf16_kernel<<<cvt_grid, 256, 0, stream>>>(w0, wbf + 0 * WSZ);
    cvt_bf16_kernel<<<cvt_grid, 256, 0, stream>>>(w1, wbf + 1 * WSZ);
    cvt_bf16_kernel<<<cvt_grid, 256, 0, stream>>>(w2, wbf + 2 * WSZ);
    cvt_bf16_kernel<<<cvt_grid, 256, 0, stream>>>(wo, wbf + 3 * WSZ);

    // 3) four mixture layers
    dim3 grid(B_TOT / BM, HDIM / BN);
    moie_layer_kernel<<<grid, 256, 0, stream>>>(x,  wbf + 0 * WSZ, b0, alpha, hA, 1);
    moie_layer_kernel<<<grid, 256, 0, stream>>>(hA, wbf + 1 * WSZ, b1, alpha, hB, 1);
    moie_layer_kernel<<<grid, 256, 0, stream>>>(hB, wbf + 2 * WSZ, b2, alpha, hA, 1);
    moie_layer_kernel<<<grid, 256, 0, stream>>>(hA, wbf + 3 * WSZ, bo, alpha,
                                                (float*)d_out, 0);
}
